// SwinTransformerBlock3D_34514357190747
// MI455X (gfx1250) — matrix operand order repaired and checked by reading, not compiled, not run on hardware
//
#include <hip/hip_runtime.h>

// ---------------------------------------------------------------------------
// Types for CDNA5 WMMA (wave32, v_wmma_f32_16x16x32_f16)
// ---------------------------------------------------------------------------
typedef _Float16 f16;
typedef __attribute__((ext_vector_type(16))) _Float16 v16h;
typedef __attribute__((ext_vector_type(8)))  _Float16 v8h;
typedef __attribute__((ext_vector_type(8)))  float    v8f;
typedef __attribute__((ext_vector_type(4)))  int      v4i;

#define WMMA_F32_F16(a, b, c) \
  __builtin_amdgcn_wmma_f32_16x16x32_f16(false, (a), false, (b), (short)0, (c), false, false)

// Async global->LDS (ASYNCcnt-tracked DMA), guarded so the file compiles on
// toolchains without the gfx1250 async builtins (falls back to sync staging).
#if defined(__has_builtin)
#if __has_builtin(__builtin_amdgcn_global_load_async_to_lds_b128) && \
    __has_builtin(__builtin_amdgcn_s_wait_asynccnt)
#define HAS_ASYNC_LDS 1
#endif
#endif
#ifndef HAS_ASYNC_LDS
#define HAS_ASYNC_LDS 0
#endif

__device__ __forceinline__ void async_cp16(const f16* g, f16* l) {
#if HAS_ASYNC_LDS
  f16* gg = const_cast<f16*>(g);
  __builtin_amdgcn_global_load_async_to_lds_b128(
      (__attribute__((address_space(1))) v4i*)gg,
      (__attribute__((address_space(3))) v4i*)l, 0, 0);
#else
  (void)g; (void)l;
#endif
}

// Problem constants (from reference)
constexpr int    Bc   = 2,  Dc = 16, Hc = 56, Wc = 56, Cc = 384;
constexpr int    NHc  = 12, HDc = 32, Nwin = 64;            // heads, head dim, tokens/window
constexpr int    WINB = 784;                                 // windows per batch: 4*14*14
constexpr int    BWc  = Bc * WINB;                           // 1568 windows
constexpr size_t Tc   = (size_t)Bc * Dc * Hc * Wc;           // 100352 tokens
constexpr size_t SA   = Tc * Cc;                             // elements of one (T,C) tensor

// ---------------------------------------------------------------------------
// Fragment loaders (LDS -> VGPR), per ISA 7.12.2 layouts.
// A-matrix 16x32 f16: lane(0-15)=M, halves j=0..7 -> K=koff..koff+7,
// j=8..15 -> K=koff+16..koff+23, koff = 8*(lane>=16).
// ---------------------------------------------------------------------------
__device__ __forceinline__ v16h ld_fragA(const f16* base, int pitch, int lane) {
  const f16* p = base + (size_t)(lane & 15) * pitch + ((lane >> 4) << 3);
  v8h lo = *reinterpret_cast<const v8h*>(p);
  v8h hi = *reinterpret_cast<const v8h*>(p + 16);
  v16h r;
#pragma unroll
  for (int t = 0; t < 8; ++t) { r[t] = lo[t]; r[t + 8] = hi[t]; }
  return r;
}

// B-matrix 32x16 f16 (loaded from B^T rows, i.e. (N,K) row-major):
// lane(0-15)=N column, j=0..7 -> K=koff+j, j=8..15 -> K=koff+16+(j-8), koff = 8*(lane>=16).
__device__ __forceinline__ v16h ld_fragB(const f16* base, int pitch, int lane) {
  const f16* p = base + (size_t)(lane & 15) * pitch + ((lane >> 4) << 3);
  v8h lo = *reinterpret_cast<const v8h*>(p);
  v8h hi = *reinterpret_cast<const v8h*>(p + 16);
  v16h r;
#pragma unroll
  for (int t = 0; t < 8; ++t) { r[t] = lo[t]; r[t + 8] = hi[t]; }
  return r;
}

__device__ __forceinline__ float gelu_exact(float v) {
  return 0.5f * v * (1.0f + erff(v * 0.70710678118654752f));
}

// ---------------------------------------------------------------------------
// Weight transpose + fp32 -> f16:  src (K,N) row-major  ->  dst (N,K) row-major
// ---------------------------------------------------------------------------
__global__ void wtr_kernel(const float* __restrict__ src, f16* __restrict__ dst,
                           int K, int N) {
  int idx = blockIdx.x * 256 + threadIdx.x;
  if (idx < K * N) {
    int k = idx / N, n = idx - k * N;
    dst[(size_t)n * K + k] = (f16)src[idx];
  }
}

// ---------------------------------------------------------------------------
// LayerNorm kernels: one wave (32 lanes) per token, 12 channels per lane.
// SHIFT=true: also performs roll(-2,-2,-2) + 4x4x4 window partition.
// ---------------------------------------------------------------------------
template <bool SHIFT>
__global__ __launch_bounds__(256) void ln_kernel(const float* __restrict__ x,
                                                 const float* __restrict__ g,
                                                 const float* __restrict__ bb,
                                                 f16* __restrict__ out) {
  int lane = threadIdx.x & 31;
  int wave = threadIdx.x >> 5;
  size_t r = (size_t)blockIdx.x * 8 + wave;   // output token row
  size_t src;
  if (SHIFT) {
    int w = (int)(r >> 6), i = (int)(r & 63);
    int b  = w / WINB;  int rem = w % WINB;
    int wd = rem / 196; int wh = (rem / 14) % 14; int ww = rem % 14;
    int d  = (wd * 4 + (i >> 4) + 2) & 15;
    int hh = (wh * 4 + ((i >> 2) & 3) + 2) % 56;
    int wc = (ww * 4 + (i & 3) + 2) % 56;
    src = (((size_t)b * Dc + d) * Hc + hh) * Wc + wc;
  } else {
    src = r;
  }
  const float* xp = x + src * Cc;
  float v[12];
  float s = 0.0f;
#pragma unroll
  for (int j = 0; j < 12; ++j) { v[j] = xp[lane + 32 * j]; s += v[j]; }
#pragma unroll
  for (int off = 1; off < 32; off <<= 1) s += __shfl_xor(s, off);
  float mu = s * (1.0f / 384.0f);
  float var = 0.0f;
#pragma unroll
  for (int j = 0; j < 12; ++j) { float d0 = v[j] - mu; var += d0 * d0; }
#pragma unroll
  for (int off = 1; off < 32; off <<= 1) var += __shfl_xor(var, off);
  float rs = rsqrtf(var * (1.0f / 384.0f) + 1e-5f);
  f16* op = out + r * Cc;
#pragma unroll
  for (int j = 0; j < 12; ++j) {
    int c = lane + 32 * j;
    op[c] = (f16)((v[j] - mu) * rs * g[c] + bb[c]);
  }
}

// ---------------------------------------------------------------------------
// Tiled WMMA GEMM:  C[M,N] = A[M,K] (f16, row-major) @ Bt[N,K]^T (f16)
// Block tile 128x128, K-step 32, 8 waves as 4(M) x 2(N): wave tile 32x64,
// 8 WMMA per wave per K-step. Double-buffered LDS staged with
// GLOBAL_LOAD_ASYNC_TO_LDS_B128 (ASYNCcnt) when available.
// Epilogues:
//   0: bias, store f16
//   1: bias + exact GELU, store f16
//   2: bias + window-reverse + roll(+2) + residual(x) -> fp32 d_out
//   3: bias + residual(d_out) -> fp32 d_out
// ---------------------------------------------------------------------------
template <int EPI>
__global__ __launch_bounds__(256) void gemm_f16(
    const f16* __restrict__ A, const f16* __restrict__ Bt,
    const float* __restrict__ bias, f16* __restrict__ Of16,
    float* __restrict__ Of32, const float* __restrict__ resid,
    int M, int N, int K) {
  (void)M;
  constexpr int PAD = 40;                       // 32 + 8 halves: conflict-free pitch
  __shared__ __align__(16) f16 sA[2][128 * PAD];
  __shared__ __align__(16) f16 sB[2][128 * PAD];

  int tid  = threadIdx.x;
  int lane = tid & 31, wave = tid >> 5;
  int wm = wave & 3, wn = wave >> 2;            // wave tile = 32x64
  const f16* Ab = A  + (size_t)blockIdx.x * 128 * K;
  const f16* Bb = Bt + (size_t)blockIdx.y * 128 * K;

  // Per-thread staging slots: rows r0 and r0+64, 8-half chunk kc0.
  int r0  = tid >> 2;
  int kc0 = (tid & 3) * 8;

#if HAS_ASYNC_LDS
  auto issue = [&](int kb, int buf) {
    int ko = kb * 32 + kc0;
    async_cp16(&Ab[(size_t)r0 * K + ko],        &sA[buf][r0 * PAD + kc0]);
    async_cp16(&Ab[(size_t)(r0 + 64) * K + ko], &sA[buf][(r0 + 64) * PAD + kc0]);
    async_cp16(&Bb[(size_t)r0 * K + ko],        &sB[buf][r0 * PAD + kc0]);
    async_cp16(&Bb[(size_t)(r0 + 64) * K + ko], &sB[buf][(r0 + 64) * PAD + kc0]);
  };
#else
  auto stage_sync = [&](int kb, int buf) {
    int ko = kb * 32 + kc0;
    *reinterpret_cast<v8h*>(&sA[buf][r0 * PAD + kc0]) =
        *reinterpret_cast<const v8h*>(&Ab[(size_t)r0 * K + ko]);
    *reinterpret_cast<v8h*>(&sA[buf][(r0 + 64) * PAD + kc0]) =
        *reinterpret_cast<const v8h*>(&Ab[(size_t)(r0 + 64) * K + ko]);
    *reinterpret_cast<v8h*>(&sB[buf][r0 * PAD + kc0]) =
        *reinterpret_cast<const v8h*>(&Bb[(size_t)r0 * K + ko]);
    *reinterpret_cast<v8h*>(&sB[buf][(r0 + 64) * PAD + kc0]) =
        *reinterpret_cast<const v8h*>(&Bb[(size_t)(r0 + 64) * K + ko]);
  };
#endif

  v8f acc[2][4] = {};
  const int nK = K >> 5;

#if HAS_ASYNC_LDS
  issue(0, 0);
#endif
  for (int kb = 0; kb < nK; ++kb) {
    int cur = kb & 1;
#if HAS_ASYNC_LDS
    if (kb + 1 < nK) {
      issue(kb + 1, cur ^ 1);
      // wait for stage kb's 4 per-thread transfers (stage kb+1 stays in flight)
      __builtin_amdgcn_s_wait_asynccnt(4);
    } else {
      __builtin_amdgcn_s_wait_asynccnt(0);
    }
    __syncthreads();
#else
    stage_sync(kb, cur);
    if (kb + 1 < nK) {  // warm L2 for the next K-slice (global_prefetch_b8)
      __builtin_prefetch(&Ab[(size_t)(tid & 127) * K + (kb + 1) * 32], 0, 1);
      if (tid < 128) __builtin_prefetch(&Bb[(size_t)tid * K + (kb + 1) * 32], 0, 1);
    }
    __syncthreads();
#endif

    v16h af0 = ld_fragA(&sA[cur][(wm * 32 + 0)  * PAD], PAD, lane);
    v16h af1 = ld_fragA(&sA[cur][(wm * 32 + 16) * PAD], PAD, lane);
#pragma unroll
    for (int tj = 0; tj < 4; ++tj) {
      v16h bf = ld_fragB(&sB[cur][(wn * 64 + tj * 16) * PAD], PAD, lane);
      acc[0][tj] = WMMA_F32_F16(af0, bf, acc[0][tj]);
      acc[1][tj] = WMMA_F32_F16(af1, bf, acc[1][tj]);
    }
    __syncthreads();
  }

  // Epilogue. D layout: vgpr r -> M = r + 8*(lane>=16); N = lane&15.
  int mBase = blockIdx.x * 128 + wm * 32;
  int nBase = blockIdx.y * 128 + wn * 64;
#pragma unroll
  for (int ti = 0; ti < 2; ++ti) {
#pragma unroll
    for (int tj = 0; tj < 4; ++tj) {
#pragma unroll
      for (int rr = 0; rr < 8; ++rr) {
        int m = mBase + ti * 16 + rr + ((lane >> 4) << 3);
        int n = nBase + tj * 16 + (lane & 15);
        float v = acc[ti][tj][rr] + bias[n];
        if (EPI == 0) {
          Of16[(size_t)m * N + n] = (f16)v;
        } else if (EPI == 1) {
          Of16[(size_t)m * N + n] = (f16)gelu_exact(v);
        } else if (EPI == 2) {
          // m is a window-order row: scatter through window-reverse + roll(+2)
          int w = m >> 6, i = m & 63;
          int b  = w / WINB;  int rem = w % WINB;
          int wd = rem / 196; int wh = (rem / 14) % 14; int ww = rem % 14;
          int d  = (wd * 4 + (i >> 4) + 2) & 15;
          int hh = (wh * 4 + ((i >> 2) & 3) + 2) % 56;
          int wc = (ww * 4 + (i & 3) + 2) % 56;
          size_t dst = (((size_t)b * Dc + d) * Hc + hh) * Wc + wc;
          Of32[dst * Cc + n] = resid[dst * Cc + n] + v;
        } else {  // EPI == 3
          size_t o = (size_t)m * N + n;
          Of32[o] = resid[o] + v;
        }
      }
    }
  }
}

// ---------------------------------------------------------------------------
// Windowed attention: one block (4 waves) per (window, head).
// N=64 tokens, hd=32 -> QK^T tile is exactly ONE v_wmma_f32_16x16x32_f16.
// Softmax fully in registers: a row (64 vals) lives in one 16-lane half-wave
// across 4 accumulators -> shfl_xor(1,2,4,8) reductions.
// ---------------------------------------------------------------------------
__global__ __launch_bounds__(128) void attn_kernel(const f16* __restrict__ qkv,
                                                   const float* __restrict__ rpb,
                                                   f16* __restrict__ out) {
  constexpr int QP = 40, VP = 72, PP = 72;
  __shared__ __align__(16) f16 sQ[Nwin * QP];
  __shared__ __align__(16) f16 sK[Nwin * QP];
  __shared__ __align__(16) f16 sVt[HDc * VP];   // V^T : 32 x 64
  __shared__ __align__(16) f16 sP[Nwin * PP];   // softmax(P) : 64 x 64
  __shared__ int sCnt[Nwin];

  int w    = blockIdx.x;        // window
  int head = blockIdx.y;        // head
  int tid  = threadIdx.x;
  int lane = tid & 31, wv = tid >> 5;

  // region counts for the shift mask (computed on the fly; Dp=16,Hp=Wp=56)
  if (tid < Nwin) {
    int rem = w % WINB;
    int wd = rem / 196, wh = (rem / 14) % 14, ww = rem % 14;
    int d  = wd * 4 + (tid >> 4);
    int hh = wh * 4 + ((tid >> 2) & 3);
    int wc = ww * 4 + (tid & 3);
    int rd = d  < 12 ? 0 : (d  < 14 ? 1 : 2);
    int rh = hh < 52 ? 0 : (hh < 54 ? 1 : 2);
    int rw = wc < 52 ? 0 : (wc < 54 ? 1 : 2);
    sCnt[tid] = rd * 9 + rh * 3 + rw;
  }

  // Stage Q,K (row-major 64x32) and V transposed (32x64)
  size_t rowBase = (size_t)w * 64;
#pragma unroll
  for (int rep = 0; rep < 2; ++rep) {
    int cch = tid + rep * 128;              // 0..255
    int n = cch >> 2, kc = cch & 3;
    const f16* qp = &qkv[(rowBase + n) * 1152 + head * 32 + kc * 8];
    *reinterpret_cast<v8h*>(&sQ[n * QP + kc * 8]) = *reinterpret_cast<const v8h*>(qp);
    *reinterpret_cast<v8h*>(&sK[n * QP + kc * 8]) = *reinterpret_cast<const v8h*>(qp + 384);
    v8h vv = *reinterpret_cast<const v8h*>(qp + 768);
#pragma unroll
    for (int t = 0; t < 8; ++t) sVt[(kc * 8 + t) * VP + n] = vv[t];
  }
  __syncthreads();

  // --- S = scale * Q @ K^T  (each wave: 16x64 strip, 4 WMMAs) ---
  v16h aq = ld_fragA(&sQ[(wv * 16) * QP], QP, lane);
  float vals[4][8];
  const float scale = 0.17677669529663688f;   // 32^-0.5
#pragma unroll
  for (int tn = 0; tn < 4; ++tn) {
    v16h bk = ld_fragB(&sK[(tn * 16) * QP], QP, lane);
    v8f z = {};
    v8f s = WMMA_F32_F16(aq, bk, z);
#pragma unroll
    for (int rr = 0; rr < 8; ++rr) {
      int i = wv * 16 + rr + ((lane >> 4) << 3);
      int j = tn * 16 + (lane & 15);
      int ridx = (((i >> 4) - (j >> 4) + 3) * 7 +
                  (((i >> 2) & 3) - ((j >> 2) & 3) + 3)) * 7 +
                 ((i & 3) - (j & 3) + 3);
      float bb = rpb[ridx * NHc + head];
      float mm = (sCnt[i] == sCnt[j]) ? 0.0f : -100.0f;
      vals[tn][rr] = s[rr] * scale + bb + mm;
    }
  }

  // --- softmax over j (16 lanes x 4 accumulators per row) ---
#pragma unroll
  for (int rr = 0; rr < 8; ++rr) {
    float m = vals[0][rr];
#pragma unroll
    for (int tn = 1; tn < 4; ++tn) m = fmaxf(m, vals[tn][rr]);
#pragma unroll
    for (int off = 1; off < 16; off <<= 1) m = fmaxf(m, __shfl_xor(m, off));
    float sum = 0.0f;
#pragma unroll
    for (int tn = 0; tn < 4; ++tn) {
      float e = __expf(vals[tn][rr] - m);
      vals[tn][rr] = e;
      sum += e;
    }
#pragma unroll
    for (int off = 1; off < 16; off <<= 1) sum += __shfl_xor(sum, off);
    float inv = 1.0f / sum;
#pragma unroll
    for (int tn = 0; tn < 4; ++tn) vals[tn][rr] *= inv;
  }

  // spill P to LDS as f16 for the P@V WMMAs
#pragma unroll
  for (int tn = 0; tn < 4; ++tn)
#pragma unroll
    for (int rr = 0; rr < 8; ++rr) {
      int i = wv * 16 + rr + ((lane >> 4) << 3);
      int j = tn * 16 + (lane & 15);
      sP[i * PP + j] = (f16)vals[tn][rr];
    }
  __syncthreads();

  // --- O = P @ V  (each wave 16x32 strip, K=64 -> 2 k-steps x 2 col tiles) ---
  v8f o[2] = {};
#pragma unroll
  for (int ks = 0; ks < 2; ++ks) {
    v16h ap = ld_fragA(&sP[(wv * 16) * PP + ks * 32], PP, lane);
#pragma unroll
    for (int tn = 0; tn < 2; ++tn) {
      v16h bv = ld_fragB(&sVt[(tn * 16) * VP + ks * 32], VP, lane);
      o[tn] = WMMA_F32_F16(ap, bv, o[tn]);
    }
  }
#pragma unroll
  for (int tn = 0; tn < 2; ++tn)
#pragma unroll
    for (int rr = 0; rr < 8; ++rr) {
      int i = wv * 16 + rr + ((lane >> 4) << 3);
      int dcol = tn * 16 + (lane & 15);
      out[(rowBase + i) * Cc + head * 32 + dcol] = (f16)o[tn][rr];
    }
}

// ---------------------------------------------------------------------------
// Host-side orchestration
// ---------------------------------------------------------------------------
extern "C" void kernel_launch(void* const* d_in, const int* in_sizes, int n_in,
                              void* d_out, int out_size, void* d_ws, size_t ws_size,
                              hipStream_t stream) {
  (void)in_sizes; (void)n_in; (void)out_size; (void)ws_size;
  const float* x      = (const float*)d_in[0];
  const float* n1g    = (const float*)d_in[1];
  const float* n1b    = (const float*)d_in[2];
  const float* qkv_w  = (const float*)d_in[3];
  const float* qkv_b  = (const float*)d_in[4];
  const float* rpb    = (const float*)d_in[5];
  const float* proj_w = (const float*)d_in[6];
  const float* proj_b = (const float*)d_in[7];
  const float* n2g    = (const float*)d_in[8];
  const float* n2b    = (const float*)d_in[9];
  const float* fc1_w  = (const float*)d_in[10];
  const float* fc1_b  = (const float*)d_in[11];
  const float* fc2_w  = (const float*)d_in[12];
  const float* fc2_b  = (const float*)d_in[13];
  float* outp = (float*)d_out;

  // Workspace layout (halves). Regions are reused once their producer's
  // consumer has finished: attn_out overwrites win; fc1out overwrites win+qkv.
  f16* win   = (f16*)d_ws;                 // [0, SA)         LN1+shift, window order
  f16* qkvb  = win + SA;                   // [SA, 4SA)       QKV f16
  f16* h2    = qkvb + 3 * SA;              // [4SA, 5SA)      LN2 out
  f16* wq_t  = h2 + SA;                    // 1152x384
  f16* wp_t  = wq_t + (size_t)384 * 1152;  // 384x384
  f16* w1_t  = wp_t + (size_t)384 * 384;   // 1536x384
  f16* w2_t  = w1_t + (size_t)384 * 1536;  // 384x1536
  f16* attno = win;                        // reuse [0, SA)
  f16* fc1o  = (f16*)d_ws;                 // reuse [0, 4SA) == 100352*1536 halves

  // 1) weights -> f16, transposed to (N,K)
  wtr_kernel<<<(384 * 1152 + 255) / 256, 256, 0, stream>>>(qkv_w, wq_t, 384, 1152);
  wtr_kernel<<<(384 * 384  + 255) / 256, 256, 0, stream>>>(proj_w, wp_t, 384, 384);
  wtr_kernel<<<(384 * 1536 + 255) / 256, 256, 0, stream>>>(fc1_w, w1_t, 384, 1536);
  wtr_kernel<<<(1536 * 384 + 255) / 256, 256, 0, stream>>>(fc2_w, w2_t, 1536, 384);

  // 2) LN1 + roll(-2) + window partition  (wave per token, 8 tokens/block)
  ln_kernel<true><<<(unsigned)(Tc / 8), 256, 0, stream>>>(x, n1g, n1b, win);

  // 3) QKV GEMM: 100352 x 1152 x 384
  gemm_f16<0><<<dim3(784, 9), 256, 0, stream>>>(win, wq_t, qkv_b, qkvb,
                                                nullptr, nullptr, (int)Tc, 1152, 384);

  // 4) attention per (window, head)
  attn_kernel<<<dim3(BWc, NHc), 128, 0, stream>>>(qkvb, rpb, attno);

  // 5) proj GEMM + window-reverse + roll(+2) + residual -> d_out (fp32)
  gemm_f16<2><<<dim3(784, 3), 256, 0, stream>>>(attno, wp_t, proj_b, nullptr,
                                                outp, x, (int)Tc, 384, 384);

  // 6) LN2 on d_out -> h2 (f16)
  ln_kernel<false><<<(unsigned)(Tc / 8), 256, 0, stream>>>(outp, n2g, n2b, h2);

  // 7) fc1 GEMM + GELU: 100352 x 1536 x 384
  gemm_f16<1><<<dim3(784, 12), 256, 0, stream>>>(h2, w1_t, fc1_b, fc1o,
                                                 nullptr, nullptr, (int)Tc, 1536, 384);

  // 8) fc2 GEMM + residual into d_out: 100352 x 384 x 1536
  gemm_f16<3><<<dim3(784, 3), 256, 0, stream>>>(fc1o, w2_t, fc2_b, nullptr,
                                                outp, outp, (int)Tc, 384, 1536);
}
